// HumanMAC_TemporalCrossAttention_6399501271564
// MI455X (gfx1250) — matrix operand order repaired and checked by reading, not compiled, not run on hardware
//
#include <hip/hip_runtime.h>

// ---------------------------------------------------------------------------
// Problem constants (from reference): b=16, T=256, N=24, C=512, H=8, d=64
// ---------------------------------------------------------------------------
#define BB   16
#define TT   256
#define NN   24
#define CC   512
#define HH   8
#define DD   64
#define BT   (BB * NN * TT)          // 98304 rows of [C]
#define BP   (BB * NN)               // 384 sequences

typedef __bf16 bf16_t;
typedef __attribute__((ext_vector_type(16))) __bf16 v16bf;
typedef __attribute__((ext_vector_type(8)))  __bf16 v8bf;
typedef __attribute__((ext_vector_type(2)))  __bf16 v2bf;
typedef __attribute__((ext_vector_type(8)))  float  v8f;

static __device__ __forceinline__ v8f wmma_bf16(v16bf a, v16bf b, v8f c) {
  // (neg_a, A, neg_b, B, c_mod, C, reuse_a, reuse_b)
  return __builtin_amdgcn_wmma_f32_16x16x32_bf16(false, a, false, b, (short)0, c,
                                                 false, false);
}

static __device__ __forceinline__ v16bf cat8(v8bf lo, v8bf hi) {
  return __builtin_shufflevector(lo, hi, 0, 1, 2, 3, 4, 5, 6, 7, 8, 9, 10, 11,
                                 12, 13, 14, 15);
}

// CDNA5 async memory->LDS copy, 16 bytes per active lane (ASYNCcnt-tracked).
// LDS operand is the byte offset within the workgroup's LDS allocation
// (= low 32 bits of the flat shared-pointer address).
static __device__ __forceinline__ void async_g2l_b128(unsigned int lds_off,
                                                      const void* gaddr) {
  asm volatile("global_load_async_to_lds_b128 %0, %1, off"
               :
               : "v"(lds_off), "v"((unsigned long long)(uintptr_t)gaddr)
               : "memory");
}

static __device__ __forceinline__ void wait_asynccnt0() {
#if __has_builtin(__builtin_amdgcn_s_wait_asynccnt)
  __builtin_amdgcn_s_wait_asynccnt(0);
#else
  asm volatile("s_wait_asynccnt 0" ::: "memory");
#endif
}

// ---------------------------------------------------------------------------
// Kernel 1: LayerNorm (fp32 stats) + bf16 cast, with (b,T,N,C)->(b*N,T,C)
// transpose folded into the source indexing. grid.y: 0 -> x->xn, 1 -> cx->cn
// ---------------------------------------------------------------------------
__global__ __launch_bounds__(256) void ln_cast_kernel(
    const float* __restrict__ x, const float* __restrict__ cx,
    const float* __restrict__ gamma, const float* __restrict__ beta,
    bf16_t* __restrict__ xn, bf16_t* __restrict__ cn) {
  const int row = blockIdx.x;            // [0, BT): row = bp*T + t
  const int bp  = row / TT;
  const int t   = row - bp * TT;
  const int bi  = bp / NN;
  const int n   = bp - bi * NN;
  const size_t src_off = (((size_t)bi * TT + t) * NN + n) * CC;

  const float* src = (blockIdx.y == 0 ? x : cx) + src_off;
  bf16_t*      dst = (blockIdx.y == 0 ? xn : cn) + (size_t)row * CC;

  const int tid = threadIdx.x;
  const float v0 = src[tid];
  const float v1 = src[tid + 256];
  float s  = v0 + v1;
  float sq = v0 * v0 + v1 * v1;

  __shared__ float redS[8], redQ[8];
#pragma unroll
  for (int m = 16; m >= 1; m >>= 1) {
    s  += __shfl_xor(s, m);
    sq += __shfl_xor(sq, m);
  }
  if ((tid & 31) == 0) { redS[tid >> 5] = s; redQ[tid >> 5] = sq; }
  __syncthreads();
  s = 0.f; sq = 0.f;
#pragma unroll
  for (int w = 0; w < 8; ++w) { s += redS[w]; sq += redQ[w]; }

  const float mu  = s * (1.0f / CC);
  const float var = sq * (1.0f / CC) - mu * mu;
  const float inv = rsqrtf(var + 1e-5f);

  dst[tid]       = (bf16_t)((v0 - mu) * inv * gamma[tid]       + beta[tid]);
  dst[tid + 256] = (bf16_t)((v1 - mu) * inv * gamma[tid + 256] + beta[tid + 256]);
}

// ---------------------------------------------------------------------------
// Kernel 2: weight fp32 -> bf16 cast
// ---------------------------------------------------------------------------
__global__ __launch_bounds__(256) void wcast_kernel(const float* __restrict__ w,
                                                    bf16_t* __restrict__ o,
                                                    int nElem) {
  int i = blockIdx.x * 256 + threadIdx.x;
  if (i < nElem) o[i] = (bf16_t)w[i];
}

// ---------------------------------------------------------------------------
// Kernel 3: Out[M,512] = A[M,512] @ W^T  (W stored [out,in], so B-fragment
// columns are contiguous W rows). 8 waves/block as 2(M)x4(N); wave tile 64x64.
// K loop: 512 in steps of 32 -> 16 WMMAs per step, 256 WMMAs per wave.
// A re-reads across the 4 N-waves hit the shared WGP$ / 192MB L2.
// ---------------------------------------------------------------------------
__global__ __launch_bounds__(256) void qkv_gemm_kernel(
    const bf16_t* __restrict__ A, const bf16_t* __restrict__ W,
    bf16_t* __restrict__ Out) {
  const int lane  = threadIdx.x & 31;
  const int wave  = threadIdx.x >> 5;
  const int wm    = wave & 1;
  const int wn    = wave >> 1;
  const int mbase = blockIdx.x * 128 + wm * 64;
  const int nbase = blockIdx.y * 256 + wn * 64;
  const int r     = lane & 15;
  const int h     = lane >> 4;

  v8f acc[4][4] = {};

  for (int k0 = 0; k0 < CC; k0 += 32) {
    v16bf afr[4], bfr[4];
#pragma unroll
    for (int mt = 0; mt < 4; ++mt) {
      const bf16_t* ap = A + (size_t)(mbase + mt * 16 + r) * CC + k0;
      v8bf lo = *(const v8bf*)(ap + h * 8);        // K = h*8 + e
      v8bf hi = *(const v8bf*)(ap + 16 + h * 8);   // K = 16 + h*8 + e
      afr[mt] = cat8(lo, hi);
    }
#pragma unroll
    for (int nt = 0; nt < 4; ++nt) {
      // B[c, o] = W[o, c]; lane col o, elements K = h*8 + e, 16 + h*8 + e
      const bf16_t* bp = W + (size_t)(nbase + nt * 16 + r) * CC + k0;
      bfr[nt] = cat8(*(const v8bf*)(bp + h * 8), *(const v8bf*)(bp + 16 + h * 8));
    }
#pragma unroll
    for (int mt = 0; mt < 4; ++mt)
#pragma unroll
      for (int nt = 0; nt < 4; ++nt)
        acc[mt][nt] = wmma_bf16(afr[mt], bfr[nt], acc[mt][nt]);
  }

#pragma unroll
  for (int mt = 0; mt < 4; ++mt)
#pragma unroll
    for (int nt = 0; nt < 4; ++nt)
#pragma unroll
      for (int j = 0; j < 8; ++j)
        Out[(size_t)(mbase + mt * 16 + j + 8 * h) * CC + nbase + nt * 16 + r] =
            (bf16_t)acc[mt][nt][j];
}

// ---------------------------------------------------------------------------
// Kernel 4: flash-style attention per (sequence, head). 8 waves x 32 query
// rows. Key blocks of 32: K tile staged via GLOBAL_LOAD_ASYNC_TO_LDS_B128
// (ASYNCcnt), V staged transposed via VGPRs. Online softmax in fp32; P is
// round-tripped through per-wave LDS with interleaved key columns so each
// lane writes one packed bf16x2 per row (key r -> col 2r, key 16+r -> 2r+1,
// with Vt columns permuted identically -- softmax/PV are key-order
// invariant). Epilogue adds residual x, writes fp32 (b,T,N,C).
// ---------------------------------------------------------------------------
__global__ __launch_bounds__(256) void attn_kernel(
    const bf16_t* __restrict__ Q, const bf16_t* __restrict__ K,
    const bf16_t* __restrict__ V, const float* __restrict__ X,
    float* __restrict__ Out) {
  const int bp   = blockIdx.x;   // sequence in [0, 384)
  const int head = blockIdx.y;   // [0, 8)
  const int tid  = threadIdx.x;
  const int lane = tid & 31;
  const int wave = tid >> 5;
  const int r    = lane & 15;
  const int h    = lane >> 4;

  __shared__ bf16_t Kb[32][64];       //  4 KB: keys row-major (async-filled)
  __shared__ bf16_t Vt[64][32];       //  4 KB: V transposed, permuted cols
  __shared__ bf16_t Pb[8][32][32];    // 16 KB: per-wave P tiles

  const size_t rowbase = (size_t)bp * TT;
  const int    cbase   = head * DD;

  // Q fragments for this wave's 32 rows (resident across the key loop)
  v16bf qf[2][2];
#pragma unroll
  for (int mi = 0; mi < 2; ++mi)
#pragma unroll
    for (int ks = 0; ks < 2; ++ks) {
      const bf16_t* qp =
          Q + (rowbase + wave * 32 + mi * 16 + r) * CC + cbase + ks * 32;
      qf[mi][ks] = cat8(*(const v8bf*)(qp + h * 8),
                        *(const v8bf*)(qp + 16 + h * 8));
    }

  v8f   acc[2][4] = {};
  float mrow[2][8], lrow[2][8];
#pragma unroll
  for (int mi = 0; mi < 2; ++mi)
#pragma unroll
    for (int j = 0; j < 8; ++j) { mrow[mi][j] = -1e30f; lrow[mi][j] = 0.f; }

  // staging assignment: one key row slice of 8 dims per thread
  const int key  = tid >> 3;                        // 0..31
  const int dc   = (tid & 7) * 8;                   // 0..56
  const int vcol = ((key & 15) << 1) | (key >> 4);  // permuted Vt column

  for (int kb = 0; kb < TT; kb += 32) {
    __syncthreads();   // previous iteration done reading Kb/Vt/Pb
    // K tile: memory -> LDS directly (no VGPR round-trip), 16B per lane
    async_g2l_b128((unsigned int)(uintptr_t)&Kb[key][dc],
                   K + (rowbase + kb + key) * CC + cbase + dc);
    // V tile: needs transpose, go through VGPRs
    v8bf vv = *(const v8bf*)(V + (rowbase + kb + key) * CC + cbase + dc);
#pragma unroll
    for (int e = 0; e < 8; ++e) Vt[dc + e][vcol] = vv[e];
    wait_asynccnt0();
    __syncthreads();

    // S = (Q K^T) * scale   (2 row-tiles x 2 key-tiles, K-dim = 64)
    v8f s[2][2] = {};
#pragma unroll
    for (int ni = 0; ni < 2; ++ni)
#pragma unroll
      for (int ks = 0; ks < 2; ++ks) {
        // B[c,n] = K[key n, c]: lane col = key ni*16+r, elems c = h*8+e, 16+h*8+e
        v16bf bk = cat8(*(const v8bf*)&Kb[ni * 16 + r][ks * 32 + h * 8],
                        *(const v8bf*)&Kb[ni * 16 + r][ks * 32 + 16 + h * 8]);
#pragma unroll
        for (int mi = 0; mi < 2; ++mi)
          s[mi][ni] = wmma_bf16(qf[mi][ks], bk, s[mi][ni]);
      }

    // online softmax (row = j + 8*h of tile mi; reduce across 16-lane half)
#pragma unroll
    for (int mi = 0; mi < 2; ++mi)
#pragma unroll
      for (int j = 0; j < 8; ++j) {
        float s0 = s[mi][0][j] * 0.125f;   // 1/sqrt(64)
        float s1 = s[mi][1][j] * 0.125f;
        float mn = fmaxf(s0, s1);
        mn = fmaxf(mn, __shfl_xor(mn, 1));
        mn = fmaxf(mn, __shfl_xor(mn, 2));
        mn = fmaxf(mn, __shfl_xor(mn, 4));
        mn = fmaxf(mn, __shfl_xor(mn, 8));
        mn = fmaxf(mn, mrow[mi][j]);
        const float alpha = __expf(mrow[mi][j] - mn);
        mrow[mi][j] = mn;
        const float p0 = __expf(s0 - mn);
        const float p1 = __expf(s1 - mn);
        s[mi][0][j] = p0;
        s[mi][1][j] = p1;
        float rs = p0 + p1;
        rs += __shfl_xor(rs, 1);
        rs += __shfl_xor(rs, 2);
        rs += __shfl_xor(rs, 4);
        rs += __shfl_xor(rs, 8);
        lrow[mi][j] = lrow[mi][j] * alpha + rs;
#pragma unroll
        for (int nd = 0; nd < 4; ++nd) acc[mi][nd][j] *= alpha;
      }

    // P (C-layout) -> LDS with interleaved key columns: one packed bf16x2
    // (ds_store_b32) per row instead of two scalar b16 stores. DS ops are
    // in-order within a wave; Pb is wave-private between barriers.
#pragma unroll
    for (int mi = 0; mi < 2; ++mi)
#pragma unroll
      for (int j = 0; j < 8; ++j) {
        v2bf pp;
        pp[0] = (bf16_t)s[mi][0][j];   // key r      -> col 2r
        pp[1] = (bf16_t)s[mi][1][j];   // key 16 + r -> col 2r+1
        *(v2bf*)&Pb[wave][mi * 16 + j + 8 * h][2 * r] = pp;
      }

    v16bf pf[2];
#pragma unroll
    for (int mi = 0; mi < 2; ++mi)
      pf[mi] = cat8(*(const v8bf*)&Pb[wave][mi * 16 + r][h * 8],
                    *(const v8bf*)&Pb[wave][mi * 16 + r][16 + h * 8]);

#pragma unroll
    for (int nd = 0; nd < 4; ++nd) {
      // B[k, dim]: lane col = dim nd*16+r, elems = permuted keys h*8+e, 16+h*8+e,
      // matching the column permutation used when filling Vt.
      v16bf bv = cat8(*(const v8bf*)&Vt[nd * 16 + r][h * 8],
                      *(const v8bf*)&Vt[nd * 16 + r][16 + h * 8]);
#pragma unroll
      for (int mi = 0; mi < 2; ++mi)
        acc[mi][nd] = wmma_bf16(pf[mi], bv, acc[mi][nd]);
    }
  }

  // epilogue: normalize, add residual, write (b,T,N,C) fp32
  const int bi = bp / NN;
  const int n  = bp - bi * NN;
#pragma unroll
  for (int mi = 0; mi < 2; ++mi)
#pragma unroll
    for (int j = 0; j < 8; ++j) {
      const float inv = 1.0f / lrow[mi][j];
      const int   t   = wave * 32 + mi * 16 + j + 8 * h;
      const size_t base = (((size_t)bi * TT + t) * NN + n) * CC + cbase;
#pragma unroll
      for (int nd = 0; nd < 4; ++nd) {
        const int col = nd * 16 + r;
        Out[base + col] = X[base + col] + acc[mi][nd][j] * inv;
      }
    }
}

// ---------------------------------------------------------------------------
// Host launcher
// ---------------------------------------------------------------------------
extern "C" void kernel_launch(void* const* d_in, const int* in_sizes, int n_in,
                              void* d_out, int out_size, void* d_ws,
                              size_t ws_size, hipStream_t stream) {
  const float* x     = (const float*)d_in[0];
  const float* cx    = (const float*)d_in[1];
  const float* gamma = (const float*)d_in[2];
  const float* beta  = (const float*)d_in[3];
  const float* Wq    = (const float*)d_in[4];
  const float* Wk    = (const float*)d_in[5];
  const float* Wv    = (const float*)d_in[6];
  float*       out   = (float*)d_out;

  const size_t RC = (size_t)BT * CC;   // rows*channels
  const size_t WC = (size_t)CC * CC;

  bf16_t* xn = (bf16_t*)d_ws;
  bf16_t* cn = xn + RC;
  bf16_t* wq = cn + RC;
  bf16_t* wk = wq + WC;
  bf16_t* wv = wk + WC;
  bf16_t* q  = wv + WC;
  bf16_t* k  = q + RC;
  bf16_t* v  = k + RC;

  // 1) LayerNorm + bf16 cast (both inputs)
  ln_cast_kernel<<<dim3(BT, 2), 256, 0, stream>>>(x, cx, gamma, beta, xn, cn);

  // 2) weights -> bf16
  const int wgrid = (int)((WC + 255) / 256);
  wcast_kernel<<<wgrid, 256, 0, stream>>>(Wq, wq, (int)WC);
  wcast_kernel<<<wgrid, 256, 0, stream>>>(Wk, wk, (int)WC);
  wcast_kernel<<<wgrid, 256, 0, stream>>>(Wv, wv, (int)WC);

  // 3) QKV projections (three [98304 x 512 x 512] bf16 WMMA GEMMs)
  dim3 ggrid(BT / 128, CC / 256);
  qkv_gemm_kernel<<<ggrid, 256, 0, stream>>>(xn, wq, q);
  qkv_gemm_kernel<<<ggrid, 256, 0, stream>>>(cn, wk, k);
  qkv_gemm_kernel<<<ggrid, 256, 0, stream>>>(cn, wv, v);

  // 4) attention + residual + output transpose
  attn_kernel<<<dim3(BP, HH), 256, 0, stream>>>(q, k, v, x, out);
}
